// CausalSelfAttention_72129680769245
// MI455X (gfx1250) — matrix operand order repaired and checked by reading, not compiled, not run on hardware
//
#include <hip/hip_runtime.h>

// ---- Problem constants (match reference) ----
constexpr int B_  = 4;
constexpr int T_  = 2048;
constexpr int D_  = 1024;
constexpr int H_  = 16;
constexpr int DK_ = 64;

typedef __attribute__((ext_vector_type(16))) __bf16    v16bf;
typedef __attribute__((ext_vector_type(8)))  __bf16    v8bf;
typedef __attribute__((ext_vector_type(4)))  __bf16    v4bf;
typedef __attribute__((ext_vector_type(8)))  float     v8f;
typedef __attribute__((ext_vector_type(4)))  unsigned  v4u;
typedef __attribute__((ext_vector_type(8)))  unsigned  v8u;

static __device__ inline v8f v8f_zero() {
  v8f z;
  #pragma unroll
  for (int i = 0; i < 8; ++i) z[i] = 0.0f;
  return z;
}

static __device__ inline v8f wmma_bf16(v16bf a, v16bf b, v8f c) {
  return __builtin_amdgcn_wmma_f32_16x16x32_bf16(
      false, a, false, b, (short)0, c, false, false);
}

static __device__ inline v16bf cat8(v8bf lo, v8bf hi) {
  return __builtin_shufflevector(lo, hi, 0, 1, 2, 3, 4, 5, 6, 7,
                                 8, 9, 10, 11, 12, 13, 14, 15);
}

// ------------------------------------------------------------
// TDM: 2D tile copy global(bf16) -> LDS, with LDS row padding.
//  D# group0/group1 built in SGPRs; 2-group form (VADDR2/3 NULL).
//  tile row = tile_w bf16; LDS pad appended per row via pad fields.
// ------------------------------------------------------------
static __device__ inline void tdm_load_2d_bf16(unsigned lds_addr, const void* gaddr,
                                               unsigned tensor_w, unsigned tensor_h,
                                               unsigned tile_w, unsigned tile_h,
                                               unsigned stride_elems,
                                               unsigned pad_interval_code,
                                               unsigned pad_amount_code) {
  unsigned long long ga = (unsigned long long)gaddr;
  v4u g0;
  g0[0] = 1u;                                          // count=1, user mode
  g0[1] = lds_addr;                                    // LDS byte address
  g0[2] = (unsigned)(ga & 0xFFFFFFFFu);                // global addr [31:0]
  g0[3] = (unsigned)((ga >> 32) & 0x01FFFFFFu) | (2u << 30);  // addr[56:32], type=2
  v8u g1;
  g1[0] = (1u << 16)                                   // data_size = 1 (2 bytes)
        | (1u << 20)                                   // pad_enable
        | (pad_interval_code << 22)
        | (pad_amount_code << 25);
  g1[1] = (tensor_w & 0xFFFFu) << 16;                  // tensor_dim0[15:0] @ bits 63:48
  g1[2] = (tensor_w >> 16) | ((tensor_h & 0xFFFFu) << 16);
  g1[3] = (tensor_h >> 16) | (tile_w << 16);           // tile_dim0
  g1[4] = tile_h & 0xFFFFu;                            // tile_dim1 (tile_dim2=0)
  g1[5] = stride_elems;                                // tensor_dim0_stride[31:0]
  g1[6] = 0;
  g1[7] = 0;
  asm volatile("tensor_load_to_lds %0, %1" :: "s"(g0), "s"(g1) : "memory");
}

// ============================================================
// Pre-pass 1: f32 -> bf16 (vectorized x4)
// ============================================================
__global__ __launch_bounds__(256)
void to_bf16_x4(const float4* __restrict__ in, __bf16* __restrict__ out, size_t n4) {
  size_t i = (size_t)blockIdx.x * blockDim.x + threadIdx.x;
  if (i >= n4) return;
  float4 v = in[i];
  v4bf o;
  o[0] = (__bf16)v.x; o[1] = (__bf16)v.y; o[2] = (__bf16)v.z; o[3] = (__bf16)v.w;
  *(v4bf*)&out[i * 4] = o;
}

// ============================================================
// Pre-pass 2: transpose-convert  in (K,N) f32 -> out (N,K) bf16
// ============================================================
__global__ __launch_bounds__(256)
void transpose_to_bf16(const float* __restrict__ in, __bf16* __restrict__ out,
                       int K, int N) {
  size_t idx = (size_t)blockIdx.x * blockDim.x + threadIdx.x;
  int n = (int)(idx % N);
  int k = (int)(idx / N);
  out[(size_t)n * K + k] = (__bf16)in[idx];
}

// ============================================================
// WMMA GEMM:  Out(M,N) = A(M,K)bf16 * Bt(N,K)bf16^T + bias
// Block tile 128x64, K-step 32; tiles staged via TDM; fragments
// read as 2x ds_load_b128 each (LDS row stride 80B, 16B aligned).
// ============================================================
__global__ __launch_bounds__(256)
void wmma_gemm_tdm(const __bf16* __restrict__ A, const __bf16* __restrict__ Bt,
                   const float* __restrict__ bias, float* __restrict__ Out,
                   int M, int N, int K) {
  constexpr int BM = 128, BN = 64, BK = 32, LDK = BK + 8;  // 40 bf16 = 80 B rows
  __shared__ alignas(16) __bf16 As[BM][LDK];
  __shared__ alignas(16) __bf16 Bs[BN][LDK];

  const int tid  = threadIdx.x;
  const int lane = tid & 31;
  const int w    = tid >> 5;
  const int wm   = w & 3;
  const int wn   = w >> 2;
  const int lm   = lane & 15;
  const int lh   = lane >> 4;
  const int bm0  = blockIdx.y * BM;
  const int bn0  = blockIdx.x * BN;

  const unsigned as_base = (unsigned)(unsigned long long)(void*)&As[0][0];
  const unsigned bs_base = (unsigned)(unsigned long long)(void*)&Bs[0][0];

  v8f acc[2][2];
  #pragma unroll
  for (int i = 0; i < 2; ++i)
    #pragma unroll
    for (int j = 0; j < 2; ++j) acc[i][j] = v8f_zero();

  for (int k0 = 0; k0 < K; k0 += BK) {
    if (w == 0) {
      // row = 32 bf16 = 16 dwords (code 3); pad = 8 bf16 = 4 dwords (code 3)
      tdm_load_2d_bf16(as_base, A + (size_t)bm0 * K + k0,
                       (unsigned)K, (unsigned)(M - bm0), BK, BM, (unsigned)K, 3u, 3u);
      tdm_load_2d_bf16(bs_base, Bt + (size_t)bn0 * K + k0,
                       (unsigned)K, (unsigned)(N - bn0), BK, BN, (unsigned)K, 3u, 3u);
      __builtin_amdgcn_s_wait_tensorcnt(0);
    }
    __syncthreads();

    v16bf af[2], bfv[2];
    #pragma unroll
    for (int t = 0; t < 2; ++t) {
      int row = wm * 32 + t * 16 + lm;
      const __bf16* ap = &As[row][lh * 8];             // kb = lh*8
      af[t] = cat8(*(const v8bf*)ap, *(const v8bf*)(ap + 16));
    }
    #pragma unroll
    for (int t = 0; t < 2; ++t) {
      int col = wn * 32 + t * 16 + lm;
      const __bf16* bp = &Bs[col][lh * 8];             // kb = lh*8
      bfv[t] = cat8(*(const v8bf*)bp, *(const v8bf*)(bp + 16));
    }
    #pragma unroll
    for (int ti = 0; ti < 2; ++ti)
      #pragma unroll
      for (int tj = 0; tj < 2; ++tj)
        acc[ti][tj] = wmma_bf16(af[ti], bfv[tj], acc[ti][tj]);
    __syncthreads();
  }

  #pragma unroll
  for (int ti = 0; ti < 2; ++ti)
    #pragma unroll
    for (int tj = 0; tj < 2; ++tj) {
      int gn = bn0 + wn * 32 + tj * 16 + lm;
      float bv = bias[gn];
      #pragma unroll
      for (int r = 0; r < 8; ++r) {
        int gm = bm0 + wm * 32 + ti * 16 + r + lh * 8;
        Out[(size_t)gm * N + gn] = acc[ti][tj][r] + bv;
      }
    }
}

// ============================================================
// RoPE + repack: qkv f32 (B,T,H*192) -> Q,K bf16 (B,H,T,64) roped;
// V bf16 transposed (B,H,64,T).
// ============================================================
__global__ __launch_bounds__(256)
void rope_pack(const float* __restrict__ qkv, const float* __restrict__ cosb,
               const float* __restrict__ sinb, __bf16* __restrict__ Qb,
               __bf16* __restrict__ Kb, __bf16* __restrict__ Vt) {
  size_t idx = (size_t)blockIdx.x * blockDim.x + threadIdx.x;
  int j = (int)(idx & 31);
  int t = (int)((idx >> 5) % T_);
  int h = (int)(((idx >> 5) / T_) % H_);
  int b = (int)(idx / (32ull * T_ * H_));

  const float* base = qkv + ((size_t)b * T_ + t) * (3 * D_) + h * (3 * DK_);
  float c = cosb[t * 32 + j];
  float s = sinb[t * 32 + j];

  float q1 = base[2 * j],       q2 = base[2 * j + 1];
  float k1 = base[64 + 2 * j],  k2 = base[64 + 2 * j + 1];
  float v1 = base[128 + 2 * j], v2 = base[128 + 2 * j + 1];

  size_t bh = (size_t)b * H_ + h;
  __bf16* Qp = Qb + (bh * T_ + t) * DK_;
  __bf16* Kp = Kb + (bh * T_ + t) * DK_;
  Qp[j]      = (__bf16)(q1 * c - q2 * s);
  Qp[j + 32] = (__bf16)(q1 * s + q2 * c);
  Kp[j]      = (__bf16)(k1 * c - k2 * s);
  Kp[j + 32] = (__bf16)(k1 * s + k2 * c);
  Vt[(bh * DK_ + 2 * j)     * T_ + t] = (__bf16)v1;
  Vt[(bh * DK_ + 2 * j + 1) * T_ + t] = (__bf16)v2;
}

// ============================================================
// Causal flash attention (online softmax), bf16 WMMA.
// Grid: (T/128, B*H); 8 waves/block; each wave owns 16 query rows.
// ============================================================
__global__ __launch_bounds__(256)
void flash_attn(const __bf16* __restrict__ Qb, const __bf16* __restrict__ Kb,
                const __bf16* __restrict__ Vt, __bf16* __restrict__ Oa) {
  __shared__ alignas(16) __bf16 Pl[8][16][40];  // wave-private P staging

  const int lane = threadIdx.x & 31;
  const int w    = threadIdx.x >> 5;
  const int lm   = lane & 15;
  const int lh   = lane >> 4;
  const int bh   = blockIdx.y;
  const int b    = bh / H_;
  const int h    = bh % H_;
  const int q0   = blockIdx.x * 128 + w * 16;

  const __bf16* Qp = Qb + (size_t)bh * T_ * DK_;
  const __bf16* Kp = Kb + (size_t)bh * T_ * DK_;
  const __bf16* Vp = Vt + (size_t)bh * DK_ * T_;

  v16bf aq[2];
  {
    int row = q0 + lm;
    int kb  = lh * 8;
    #pragma unroll
    for (int c = 0; c < 2; ++c)
      #pragma unroll
      for (int i = 0; i < 16; ++i)
        aq[c][i] = Qp[(size_t)row * DK_ + c * 32 + kb + i + ((i >= 8) ? 8 : 0)];
  }

  float mi[8], li[8];
  v8f acc[4];
  #pragma unroll
  for (int r = 0; r < 8; ++r) { mi[r] = -3.0e38f; li[r] = 0.0f; }
  #pragma unroll
  for (int n = 0; n < 4; ++n) acc[n] = v8f_zero();

  const float scale = 0.125f;

  for (int s0 = 0; s0 < q0 + 16; s0 += 32) {
    v8f S[2];
    #pragma unroll
    for (int half = 0; half < 2; ++half) {
      v16bf bk0, bk1;
      int key = s0 + half * 16 + lm;
      #pragma unroll
      for (int i = 0; i < 16; ++i) {
        int kk = lh * 8 + i + ((i >= 8) ? 8 : 0);
        bk0[i] = Kp[(size_t)key * DK_ + kk];
        bk1[i] = Kp[(size_t)key * DK_ + 32 + kk];
      }
      v8f s = wmma_bf16(aq[0], bk0, v8f_zero());
      s     = wmma_bf16(aq[1], bk1, s);
      S[half] = s;
    }

    float p0[8], p1[8];
    #pragma unroll
    for (int r = 0; r < 8; ++r) {
      int row  = q0 + r + lh * 8;
      float x0 = S[0][r] * scale;
      float x1 = S[1][r] * scale;
      if (s0 + lm > row)      x0 = -3.0e38f;
      if (s0 + 16 + lm > row) x1 = -3.0e38f;
      float mx = fmaxf(x0, x1);
      #pragma unroll
      for (int off = 1; off < 16; off <<= 1) mx = fmaxf(mx, __shfl_xor(mx, off));
      float mnew = fmaxf(mi[r], mx);
      p0[r] = __expf(x0 - mnew);
      p1[r] = __expf(x1 - mnew);
      float sum = p0[r] + p1[r];
      #pragma unroll
      for (int off = 1; off < 16; off <<= 1) sum += __shfl_xor(sum, off);
      float alpha = __expf(mi[r] - mnew);
      li[r] = li[r] * alpha + sum;
      mi[r] = mnew;
      #pragma unroll
      for (int n = 0; n < 4; ++n) acc[n][r] *= alpha;
    }

    #pragma unroll
    for (int r = 0; r < 8; ++r) {
      int M = r + lh * 8;
      Pl[w][M][lm]      = (__bf16)p0[r];
      Pl[w][M][16 + lm] = (__bf16)p1[r];
    }
    v16bf pf;
    {
      const __bf16* pp = &Pl[w][lm][lh * 8];
      pf = cat8(*(const v8bf*)pp, *(const v8bf*)(pp + 16));
    }

    #pragma unroll
    for (int n = 0; n < 4; ++n) {
      v16bf bv;
      int col = n * 16 + lm;
      #pragma unroll
      for (int i = 0; i < 16; ++i) {
        int kk = lh * 8 + i + ((i >= 8) ? 8 : 0);
        bv[i] = Vp[(size_t)col * T_ + s0 + kk];
      }
      acc[n] = wmma_bf16(pf, bv, acc[n]);
    }
  }

  #pragma unroll
  for (int r = 0; r < 8; ++r) {
    int row   = q0 + r + lh * 8;
    float inv = 1.0f / li[r];
    #pragma unroll
    for (int n = 0; n < 4; ++n) {
      int col = h * DK_ + n * 16 + lm;
      Oa[((size_t)b * T_ + row) * D_ + col] = (__bf16)(acc[n][r] * inv);
    }
  }
}

// ============================================================
// Launch
// ============================================================
extern "C" void kernel_launch(void* const* d_in, const int* in_sizes, int n_in,
                              void* d_out, int out_size, void* d_ws, size_t ws_size,
                              hipStream_t stream) {
  (void)in_sizes; (void)n_in; (void)out_size; (void)ws_size;

  const float* x     = (const float*)d_in[0];
  const float* W_qkv = (const float*)d_in[1];
  const float* b_qkv = (const float*)d_in[2];
  const float* W_out = (const float*)d_in[3];
  const float* b_out = (const float*)d_in[4];
  const float* cosb  = (const float*)d_in[5];
  const float* sinb  = (const float*)d_in[6];
  float* out = (float*)d_out;

  const int M  = B_ * T_;  // 8192
  const int N1 = 3 * D_;   // 3072
  const int K  = D_;       // 1024

  char* ws = (char*)d_ws;
  size_t off = 0;
  float*  qkv = (float*)(ws + off);  off += (size_t)M * N1 * sizeof(float);
  __bf16* Qb  = (__bf16*)(ws + off); off += (size_t)M * D_ * sizeof(__bf16);
  __bf16* Kb  = (__bf16*)(ws + off); off += (size_t)M * D_ * sizeof(__bf16);
  __bf16* Vt  = (__bf16*)(ws + off); off += (size_t)M * D_ * sizeof(__bf16);
  __bf16* Oa  = (__bf16*)(ws + off); off += (size_t)M * D_ * sizeof(__bf16);
  __bf16* xb  = (__bf16*)(ws + off); off += (size_t)M * K  * sizeof(__bf16);
  __bf16* WqT = (__bf16*)(ws + off); off += (size_t)N1 * K * sizeof(__bf16);
  __bf16* WoT = (__bf16*)(ws + off); off += (size_t)D_ * K * sizeof(__bf16);

  // 0) Pre-convert activations / transpose-convert weights to bf16
  {
    size_t n4 = (size_t)M * K / 4;
    to_bf16_x4<<<(int)((n4 + 255) / 256), 256, 0, stream>>>((const float4*)x, xb, n4);
    size_t nq = (size_t)K * N1;
    transpose_to_bf16<<<(int)(nq / 256), 256, 0, stream>>>(W_qkv, WqT, K, N1);
    size_t no = (size_t)K * D_;
    transpose_to_bf16<<<(int)(no / 256), 256, 0, stream>>>(W_out, WoT, K, D_);
  }

  // 1) QKV projection (TDM-staged WMMA GEMM)
  wmma_gemm_tdm<<<dim3(N1 / 64, M / 128), 256, 0, stream>>>(
      xb, WqT, b_qkv, qkv, M, N1, K);

  // 2) RoPE + repack
  {
    long long total = (long long)B_ * H_ * T_ * 32;
    rope_pack<<<(int)(total / 256), 256, 0, stream>>>(qkv, cosb, sinb, Qb, Kb, Vt);
  }

  // 3) Causal flash attention
  flash_attn<<<dim3(T_ / 128, B_ * H_), 256, 0, stream>>>(Qb, Kb, Vt, Oa);

  // 4) Output projection
  wmma_gemm_tdm<<<dim3(D_ / 64, M / 128), 256, 0, stream>>>(
      Oa, WoT, b_out, out, M, D_, K);
}
